// MLPFP8_17592186044708
// MI455X (gfx1250) — 8-bit matrix operand order repaired and checked by reading, compiled, not run on hardware
//
#include <hip/hip_runtime.h>

typedef __attribute__((ext_vector_type(16))) int   v16i;
typedef __attribute__((ext_vector_type(8)))  float v8f;
typedef __attribute__((ext_vector_type(4)))  float v4f;
typedef __attribute__((ext_vector_type(4)))  int   v4i;
typedef __attribute__((ext_vector_type(2)))  int   v2i;

#define FP8_MAX    448.0f
#define BLK_M      128
#define BLK_N      64
#define KSTEP      128
#define LDS_STRIDE 144

__device__ __forceinline__ void cp16(const void* g, void* l) {
#if defined(__AMDGCN__) && __has_builtin(__builtin_amdgcn_global_load_async_to_lds_b128)
    typedef __attribute__((address_space(1))) v4i gv4i_t;
    typedef __attribute__((address_space(3))) v4i lv4i_t;
    __builtin_amdgcn_global_load_async_to_lds_b128(
        (gv4i_t*)g, (lv4i_t*)l, 0, 0);
#else
    *reinterpret_cast<v4i*>(l) = *reinterpret_cast<const v4i*>(g);
#endif
}

__device__ __forceinline__ void wait_async_copies() {
#if defined(__AMDGCN__)
#if __has_builtin(__builtin_amdgcn_s_wait_asynccnt)
    __builtin_amdgcn_s_wait_asynccnt(0);
#else
    asm volatile("s_wait_asynccnt 0x0" ::: "memory");
#endif
#endif
}

__global__ void init_amax_kernel(unsigned* amax) {
    if (threadIdx.x < 2) amax[threadIdx.x] = 0u;
}

__global__ __launch_bounds__(256)
void amax_kernel(const float* __restrict__ p, long n, unsigned* __restrict__ out) {
    __shared__ unsigned sred[256];
    const v4f* p4 = (const v4f*)p;
    long n4 = n >> 2;
    long i = (long)blockIdx.x * blockDim.x + threadIdx.x;
    long stride = (long)gridDim.x * blockDim.x;
    float m = 0.0f;
    for (; i < n4; i += stride) {
        v4f v = p4[i];
        m = fmaxf(m, fmaxf(fmaxf(fabsf(v.x), fabsf(v.y)),
                           fmaxf(fabsf(v.z), fabsf(v.w))));
    }
    for (long t = (n4 << 2) + (long)blockIdx.x * blockDim.x + threadIdx.x; t < n;
         t += stride)
        m = fmaxf(m, fabsf(p[t]));
    sred[threadIdx.x] = __float_as_uint(m);
    __syncthreads();
    for (int s = 128; s > 0; s >>= 1) {
        if ((int)threadIdx.x < s) {
            unsigned o = sred[threadIdx.x + s];
            if (o > sred[threadIdx.x]) sred[threadIdx.x] = o;
        }
        __syncthreads();
    }
    if (threadIdx.x == 0) atomicMax(out, sred[0]);
}

__device__ __forceinline__ unsigned f32_to_e4m3_sw(float x) {
    unsigned sign = (__float_as_uint(x) >> 31) << 7;
    float ax = fabsf(x);
    if (!(ax > 0.0f)) return sign;
    ax = fminf(ax, FP8_MAX);
    int e;
    (void)frexpf(ax, &e);
    int E = e - 1 + 7;
    unsigned bits;
    if (E <= 0) {
        int mant = (int)rintf(ax * 512.0f);
        if (mant > 8) mant = 8;
        bits = (unsigned)mant;
    } else {
        float scaled = ldexpf(ax, -(e - 1));
        int mant = (int)rintf((scaled - 1.0f) * 8.0f);
        if (mant == 8) { mant = 0; E += 1; }
        if (E > 15 || (E == 15 && mant > 6)) { E = 15; mant = 6; }
        bits = (unsigned)((E << 3) | mant);
    }
    return sign | bits;
}

__global__ __launch_bounds__(256)
void quant_kernel(const float* __restrict__ in, unsigned* __restrict__ out,
                  long n4, const unsigned* __restrict__ amax_bits) {
    float amax  = fmaxf(__uint_as_float(*amax_bits), 1e-12f);
    float scale = FP8_MAX / amax;
    const v4f* in4 = (const v4f*)in;
    long i = (long)blockIdx.x * blockDim.x + threadIdx.x;
    long stride = (long)gridDim.x * blockDim.x;
    for (; i < n4; i += stride) {
        v4f v = __builtin_nontemporal_load(in4 + i);
        float a = v.x * scale, b = v.y * scale, c = v.z * scale, d = v.w * scale;
        unsigned packed;
#if defined(__AMDGCN__) && __has_builtin(__builtin_amdgcn_cvt_pk_fp8_f32)
        int p = __builtin_amdgcn_cvt_pk_fp8_f32(a, b, 0, false);
        p     = __builtin_amdgcn_cvt_pk_fp8_f32(c, d, p, true);
        packed = (unsigned)p;
#else
        packed =  f32_to_e4m3_sw(a)
               | (f32_to_e4m3_sw(b) << 8)
               | (f32_to_e4m3_sw(c) << 16)
               | (f32_to_e4m3_sw(d) << 24);
#endif
        out[i] = packed;
    }
}

__global__ __launch_bounds__(256)
void gemm_fp8_wmma_kernel(const unsigned char* __restrict__ Xq,
                          const unsigned char* __restrict__ Wq,
                          const float* __restrict__ bias,
                          const unsigned* __restrict__ amax_bits,
                          float* __restrict__ out,
                          int M, int N, int K) {
    __shared__ unsigned char Abuf[2][BLK_M * LDS_STRIDE];
    __shared__ unsigned char Bbuf[2][BLK_N * LDS_STRIDE];

    const int tid     = threadIdx.x;
    const int lane    = tid & 31;
    const int wave    = tid >> 5;
    const int laneLo  = lane & 15;
    const int laneHi  = lane >> 4;
    const int block_m = blockIdx.x * BLK_M;
    const int block_n = blockIdx.y * BLK_N;
    const int wm      = (wave >> 1) * 32;
    const int wn      = (wave & 1) * 32;

    const unsigned char* Ag = Xq + (size_t)block_m * K;
    const unsigned char* Bg = Wq + (size_t)block_n * K;

    __builtin_prefetch(bias + block_n, 0, 2);

    auto stage = [&](int buf, int k0) {
        #pragma unroll
        for (int j = 0; j < 4; ++j) {
            int e = tid + j * 256;
            int row = e >> 3, c = e & 7;
            cp16(Ag + (size_t)row * K + k0 + c * 16,
                 &Abuf[buf][row * LDS_STRIDE + c * 16]);
        }
        #pragma unroll
        for (int j = 0; j < 2; ++j) {
            int e = tid + j * 256;
            int row = e >> 3, c = e & 7;
            cp16(Bg + (size_t)row * K + k0 + c * 16,
                 &Bbuf[buf][row * LDS_STRIDE + c * 16]);
        }
    };

    v8f acc[2][2] = {};

    stage(0, 0);
    const int nk = K / KSTEP;
    for (int ks = 0; ks < nk; ++ks) {
        wait_async_copies();
        __syncthreads();
        if (ks + 1 < nk) stage((ks + 1) & 1, (ks + 1) * KSTEP);

        const int cur = ks & 1;
        v16i afrag[2], bfrag[2];
        #pragma unroll
        for (int mt = 0; mt < 2; ++mt) {
            const unsigned char* rp =
                &Abuf[cur][(wm + mt * 16 + laneLo) * LDS_STRIDE + 16 * laneHi];
            v4i* dst = reinterpret_cast<v4i*>(&afrag[mt]);
            #pragma unroll
            for (int c = 0; c < 4; ++c)
                dst[c] = *reinterpret_cast<const v4i*>(rp + 32 * c);
        }
        #pragma unroll
        for (int nt = 0; nt < 2; ++nt) {
            const unsigned char* cp =
                &Bbuf[cur][(wn + nt * 16 + laneLo) * LDS_STRIDE + 16 * laneHi];
            v4i* dst = reinterpret_cast<v4i*>(&bfrag[nt]);
            #pragma unroll
            for (int j = 0; j < 4; ++j)
                dst[j] = *reinterpret_cast<const v4i*>(cp + 32 * j);
        }

        #pragma unroll
        for (int mt = 0; mt < 2; ++mt)
            #pragma unroll
            for (int nt = 0; nt < 2; ++nt)
                acc[mt][nt] = __builtin_amdgcn_wmma_f32_16x16x128_fp8_fp8(
                    afrag[mt], bfrag[nt], (short)0, acc[mt][nt],
                     false,  false);
    }

    float ax  = fmaxf(__uint_as_float(amax_bits[0]), 1e-12f);
    float aw  = fmaxf(__uint_as_float(amax_bits[1]), 1e-12f);
    float inv = (ax * aw) * (1.0f / (FP8_MAX * FP8_MAX));

    #pragma unroll
    for (int mt = 0; mt < 2; ++mt) {
        #pragma unroll
        for (int nt = 0; nt < 2; ++nt) {
            int n = block_n + wn + nt * 16 + laneLo;
            float bv = bias[n];
            #pragma unroll
            for (int v = 0; v < 8; ++v) {
                int m = block_m + wm + mt * 16 + 8 * laneHi + v;
                __builtin_nontemporal_store(acc[mt][nt][v] * inv + bv,
                                            &out[(size_t)m * N + n]);
            }
        }
    }
}

extern "C" void kernel_launch(void* const* d_in, const int* in_sizes, int n_in,
                              void* d_out, int out_size, void* d_ws, size_t ws_size,
                              hipStream_t stream) {
    const float* x    = (const float*)d_in[0];
    const float* w    = (const float*)d_in[1];
    const float* bias = (const float*)d_in[2];
    float*       out  = (float*)d_out;

    const int  N  = in_sizes[2];
    const int  K  = in_sizes[1] / N;
    const long M  = (long)in_sizes[0] / K;
    const long nx = (long)in_sizes[0];
    const long nw = (long)in_sizes[1];

    unsigned*      amax = (unsigned*)d_ws;
    unsigned char* xq   = (unsigned char*)d_ws + 256;
    unsigned char* wq   = xq + (size_t)M * K;

    init_amax_kernel<<<1, 32, 0, stream>>>(amax);
    amax_kernel<<<1024, 256, 0, stream>>>(x, nx, amax + 0);
    amax_kernel<<<256,  256, 0, stream>>>(w, nw, amax + 1);
    quant_kernel<<<2048, 256, 0, stream>>>(x, (unsigned*)xq, nx / 4, amax + 0);
    quant_kernel<<<256,  256, 0, stream>>>(w, (unsigned*)wq, nw / 4, amax + 1);

    dim3 grid((unsigned)(M / BLK_M), (unsigned)(N / BLK_N));
    gemm_fp8_wmma_kernel<<<grid, 256, 0, stream>>>(xq, wq, bias, amax, out,
                                                   (int)M, N, K);
}
